// StatefulDPDFNet_16k_29557964931366
// MI455X (gfx1250) — compile-verified, matrix operand order checked by reading, not run on hardware
//
#include <hip/hip_runtime.h>

typedef __attribute__((ext_vector_type(16))) _Float16 v16h;
typedef __attribute__((ext_vector_type(2)))  __fp16   h2;
typedef __attribute__((ext_vector_type(8)))  float    v8f;

#define BATCH 256

union V16 { v16h v; h2 p[8]; };

__device__ __forceinline__ float sigf(float v) { return 1.f / (1.f + __expf(-v)); }

__device__ __forceinline__ float actf(float v, int a) {
  if (a == 1) return v > 0.f ? v : 0.f;
  if (a == 2) return sigf(v);
  if (a == 3) return tanhf(v);
  return v;
}

// ---------------- generic WMMA GEMM (row-major A) ----------------
// out[m,n] = act( sum_k A[m*lda+k] * W[n*wsn + k*wsk] + bias[n] + res[m*res_sm+n] )
__global__ void k_gemm(const float* __restrict__ A, int lda,
                       const float* __restrict__ W, int wsn, int wsk,
                       const float* __restrict__ bias,
                       const float* __restrict__ res, int res_sm,
                       float* __restrict__ out, int out_sm,
                       int N, int K, int act)
{
  const int lane = threadIdx.x & 31;
  const int wv   = threadIdx.x >> 5;
  const int m0   = blockIdx.x * 16;
  const int n0   = (blockIdx.y * (blockDim.x >> 5) + wv) * 16;
  if (n0 >= N) return;
  const int ml = lane & 15, hf = lane >> 4;
  const int m  = m0 + ml;
  const int nn = n0 + ml;
  const bool wvld = nn < N;
  const int  wrow = wvld ? nn * wsn : 0;
  const int  arow = m * lda;
  v8f acc = {};
  for (int kb = 0; kb < K; kb += 32) {
    if (kb + 32 < K && wvld)
      __builtin_prefetch(&W[wrow + (kb + 32) * wsk], 0, 0);
    V16 ua, ub;
#pragma unroll
    for (int j = 0; j < 8; ++j) {
      int kk = kb + ((j >> 2) << 4) + (hf << 3) + ((j & 3) << 1);
      bool v0 = kk < K, v1 = kk + 1 < K;
      int k0 = v0 ? kk : 0;
      int k1 = v1 ? kk + 1 : 0;
      float a0 = A[arow + k0];
      float a1 = A[arow + k1];
      float b0 = W[wrow + k0 * wsk];
      float b1 = W[wrow + k1 * wsk];
      a0 = v0 ? a0 : 0.f;            a1 = v1 ? a1 : 0.f;
      b0 = (wvld && v0) ? b0 : 0.f;  b1 = (wvld && v1) ? b1 : 0.f;
      ua.p[j] = __builtin_amdgcn_cvt_pkrtz(a0, a1);
      ub.p[j] = __builtin_amdgcn_cvt_pkrtz(b0, b1);
    }
    acc = __builtin_amdgcn_wmma_f32_16x16x32_f16(false, ua.v, false, ub.v, (short)0, acc, false, false);
  }
  const int n = n0 + ml;
  if (n >= N) return;
#pragma unroll
  for (int r = 0; r < 8; ++r) {
    int mm = m0 + hf * 8 + r;
    float v = acc[r];
    if (bias) v += bias[n];
    if (res)  v += res[mm * res_sm + n];
    out[mm * out_sm + n] = actf(v, act);
  }
}

// ---------------- generic WMMA im2col conv over batch ----------------
// in: (B,Cin,Fin)  W: (Cout, Cin*KW)  out: (B,Cout,Fout)
// MODE 0: fi = f*stride + k - pad ; MODE 1: convT up2 (lhs_dilation=2, pad 1)
// DUAL=1: logical channel cc -> (ci=cc/2, t=cc&1), src = t ? in2 : in, each (B,Cin/2,Fin)
template <int KW, int MODE, int DUAL>
__global__ void k_conv(const float* __restrict__ in, const float* __restrict__ in2,
                       int Cin, int Fin,
                       const float* __restrict__ W, const float* __restrict__ bias,
                       const float* __restrict__ res,
                       float* __restrict__ out, int Cout, int Fout,
                       int stride, int pad, int act)
{
  const int lane = threadIdx.x & 31;
  const int wv   = threadIdx.x >> 5;
  const int b0   = blockIdx.x * 16;
  const int f    = blockIdx.y;
  const int n0   = wv * 16;
  if (n0 >= Cout) return;
  const int ml = lane & 15, hf = lane >> 4;
  const int b  = b0 + ml;
  const int nn = n0 + ml;
  const int Ktot  = Cin * KW;
  const int halfC = Cin >> 1;
  const bool nvld = nn < Cout;
  const int  wrow = nvld ? nn * Ktot : 0;
  v8f acc = {};
  for (int kb = 0; kb < Ktot; kb += 32) {
    V16 ua, ub;
#pragma unroll
    for (int j = 0; j < 8; ++j) {
      int kk = kb + ((j >> 2) << 4) + (hf << 3) + ((j & 3) << 1);
      float av[2], wvv[2];
#pragma unroll
      for (int q = 0; q < 2; ++q) {
        int kq = kk + q;
        bool kv = kq < Ktot;
        int kqc = kv ? kq : 0;
        int cc = kqc / KW, k = kqc - cc * KW;   // KW compile-time -> magic multiply
        int fi; bool ok;
        if (MODE == 0) { fi = f * stride + k - pad; ok = (fi >= 0 && fi < Fin); }
        else { int p = f + k - 1; ok = (p >= 0) && ((p & 1) == 0); fi = p >> 1; ok = ok && (fi < Fin); }
        int fic = ok ? fi : 0;
        float a;
        if (DUAL) { int ci = cc >> 1; const float* src = (cc & 1) ? in2 : in;
                    a = src[(b * halfC + ci) * Fin + fic]; }
        else      { a = in[(b * Cin + cc) * Fin + fic]; }
        float w = W[wrow + kqc];
        av[q]  = (ok && kv) ? a : 0.f;
        wvv[q] = (nvld && kv) ? w : 0.f;
      }
      ua.p[j] = __builtin_amdgcn_cvt_pkrtz(av[0], av[1]);
      ub.p[j] = __builtin_amdgcn_cvt_pkrtz(wvv[0], wvv[1]);
    }
    acc = __builtin_amdgcn_wmma_f32_16x16x32_f16(false, ua.v, false, ub.v, (short)0, acc, false, false);
  }
  int n = n0 + ml;
  if (n >= Cout) return;
#pragma unroll
  for (int r = 0; r < 8; ++r) {
    int bb = b0 + hf * 8 + r;
    float v = acc[r] + (bias ? bias[n] : 0.f);
    int oi = (bb * Cout + n) * Fout + f;
    if (res) v += res[oi];
    out[oi] = actf(v, act);
  }
}

// ---------------- LDS-resident sequential GRU scan (hidden=64) ----------------
// x: (B,64,F), ys: (B,F,64); iterate t = t0, t0+dt, ...
// x tile for the current step is staged into LDS with async global->LDS copies.
__global__ void k_gru_seq(const float* __restrict__ x, int F,
                          const float* __restrict__ wih, const float* __restrict__ whh,
                          const float* __restrict__ bih, const float* __restrict__ bhh,
                          float* __restrict__ ys, int t0, int dt)
{
  __shared__ float hbuf[16][64];
  __shared__ float xts[16][64];
  __shared__ float gg[2][16][192];
  const int tid = threadIdx.x;
  const int lane = tid & 31, wv = tid >> 5;
  const int ml = lane & 15, hf = lane >> 4;
  const int b0 = blockIdx.x * 16;
  for (int i = tid; i < 16 * 64; i += 128) hbuf[i >> 6][i & 63] = 0.f;
  __syncthreads();
  for (int s = 0; s < F; ++s) {
    int t = t0 + dt * s;
    // ---- async stage x[:, :, t] tile (16 rows x 64 ch) into LDS ----
#if defined(__gfx1250__)
    for (int i = tid; i < 1024; i += 128) {
      int r = i & 15, c = i >> 4;
      const float* src = &x[((b0 + r) * 64 + c) * F + t];
      unsigned ldsoff = (unsigned)(unsigned long long)(uintptr_t)&xts[r][c];
      asm volatile("global_load_async_to_lds_b32 %0, %1, off"
                   :: "v"(ldsoff), "v"((unsigned long long)(uintptr_t)src)
                   : "memory");
    }
    asm volatile("s_wait_asynccnt 0x0" ::: "memory");
#else
    for (int i = tid; i < 1024; i += 128) {
      int r = i & 15, c = i >> 4;
      xts[r][c] = x[((b0 + r) * 64 + c) * F + t];
    }
#endif
    __syncthreads();
    for (int task = wv; task < 24; task += 4) {
      int which = task >= 12 ? 1 : 0;
      int n0 = (task - (which ? 12 : 0)) << 4;
      v8f acc = {};
      for (int kb = 0; kb < 64; kb += 32) {
        V16 ua, ub;
#pragma unroll
        for (int j = 0; j < 8; ++j) {
          int kk = kb + ((j >> 2) << 4) + (hf << 3) + ((j & 3) << 1);
          float a0, a1;
          if (which == 0) { a0 = xts[ml][kk];  a1 = xts[ml][kk + 1]; }
          else            { a0 = hbuf[ml][kk]; a1 = hbuf[ml][kk + 1]; }
          const float* Wp = which ? whh : wih;
          int wb = (n0 + ml) * 64 + kk;
          ua.p[j] = __builtin_amdgcn_cvt_pkrtz(a0, a1);
          ub.p[j] = __builtin_amdgcn_cvt_pkrtz(Wp[wb], Wp[wb + 1]);
        }
        acc = __builtin_amdgcn_wmma_f32_16x16x32_f16(false, ua.v, false, ub.v, (short)0, acc, false, false);
      }
      const float* bs = which ? bhh : bih;
#pragma unroll
      for (int r = 0; r < 8; ++r)
        gg[which][hf * 8 + r][n0 + ml] = acc[r] + bs[n0 + ml];
    }
    __syncthreads();
    for (int idx = tid; idx < 16 * 64; idx += 128) {
      int r = idx >> 6, j = idx & 63;
      float ir = gg[0][r][j], iz = gg[0][r][64 + j], in_ = gg[0][r][128 + j];
      float hr = gg[1][r][j], hz = gg[1][r][64 + j], hn  = gg[1][r][128 + j];
      float rg = sigf(ir + hr), z = sigf(iz + hz);
      float nv = tanhf(in_ + rg * hn);
      float h = (1.f - z) * nv + z * hbuf[r][j];
      hbuf[r][j] = h;
      ys[((b0 + r) * F + t) * 64 + j] = h;
    }
    __syncthreads();
  }
}

// ---------------- single-step GRU cell, parametric hidden H (pow2) ----------------
// x-index = m*xms1(/mdiv path) + kk*xks2
__global__ void k_gru_cell(const float* __restrict__ x, int xmdiv, int xms1, int xms2, int xks2,
                           const float* __restrict__ h,
                           const float* __restrict__ wih, const float* __restrict__ whh,
                           const float* __restrict__ bih, const float* __restrict__ bhh,
                           float* __restrict__ ho1, float* __restrict__ ho2,
                           int H, int hs)
{
  extern __shared__ float smem[];
  float* gi = smem;
  float* gh = smem + 16 * 3 * H;
  const int tid = threadIdx.x, lane = tid & 31, wv = tid >> 5;
  const int ml = lane & 15, hf = lane >> 4;
  const int m0 = blockIdx.x * 16;
  const int ntn = (3 * H) >> 4;
  const int H3 = 3 * H;
  const int m = m0 + ml;
  const int xbase = (xmdiv == 1) ? m * xms1 : (m / xmdiv) * xms1 + (m % xmdiv) * xms2;
  const int hbase = m * H;
  for (int task = wv; task < 2 * ntn; task += 8) {
    int which = task >= ntn ? 1 : 0;
    int n0 = (task - (which ? ntn : 0)) << 4;
    v8f acc = {};
    for (int kb = 0; kb < H; kb += 32) {
      V16 ua, ub;
#pragma unroll
      for (int j = 0; j < 8; ++j) {
        int kk = kb + ((j >> 2) << 4) + (hf << 3) + ((j & 3) << 1);
        float a0, a1;
        if (which == 0) { a0 = x[xbase + kk * xks2]; a1 = x[xbase + (kk + 1) * xks2]; }
        else            { a0 = h[hbase + kk];        a1 = h[hbase + kk + 1]; }
        const float* Wp = which ? whh : wih;
        int wb = (n0 + ml) * H + kk;
        ua.p[j] = __builtin_amdgcn_cvt_pkrtz(a0, a1);
        ub.p[j] = __builtin_amdgcn_cvt_pkrtz(Wp[wb], Wp[wb + 1]);
      }
      acc = __builtin_amdgcn_wmma_f32_16x16x32_f16(false, ua.v, false, ub.v, (short)0, acc, false, false);
    }
    float* gp = which ? gh : gi;
    const float* bs = which ? bhh : bih;
#pragma unroll
    for (int r = 0; r < 8; ++r)
      gp[(hf * 8 + r) * H3 + n0 + ml] = acc[r] + bs[n0 + ml];
  }
  __syncthreads();
  for (int idx = tid; idx < (16 << hs); idx += 256) {
    int r = idx >> hs, j = idx & (H - 1);
    float ir = gi[r * H3 + j], iz = gi[r * H3 + H + j], in_ = gi[r * H3 + 2 * H + j];
    float hr = gh[r * H3 + j], hz = gh[r * H3 + H + j], hn  = gh[r * H3 + 2 * H + j];
    float rg = sigf(ir + hr), z = sigf(iz + hz);
    float nv = tanhf(in_ + rg * hn);
    int mm = m0 + r;
    float hnew = (1.f - z) * nv + z * h[mm * H + j];
    ho1[mm * H + j] = hnew;
    if (ho2) ho2[mm * H + j] = hnew;
  }
}

// ---------------- linear (N=64) + layernorm + accumulate into (B,64,F) ----------------
// K must be 64 or 128 (multiple of 32). If no second half, pass Ab = Af.
__global__ void k_lin_ln_add(const float* __restrict__ Af, const float* __restrict__ Ab, int K,
                             const float* __restrict__ W, const float* __restrict__ bias,
                             const float* __restrict__ g, const float* __restrict__ be,
                             float* __restrict__ xacc, int F)
{
  __shared__ float tile[16][64];
  __shared__ float mu[16], iv[16];
  const int tid = threadIdx.x, lane = tid & 31, wv = tid >> 5;
  const int ml = lane & 15, hf = lane >> 4;
  const int m0 = blockIdx.x * 16;
  const int n0 = wv * 16;
  {
    const int m = m0 + ml;
    v8f acc = {};
    for (int kb = 0; kb < K; kb += 32) {
      V16 ua, ub;
#pragma unroll
      for (int j = 0; j < 8; ++j) {
        int kk = kb + ((j >> 2) << 4) + (hf << 3) + ((j & 3) << 1);
        int kc = kk & 63;
        float f0 = Af[m * 64 + kc],     fb0 = Ab[m * 64 + kc];
        float f1 = Af[m * 64 + kc + 1], fb1 = Ab[m * 64 + kc + 1];
        float a0 = (kk < 64) ? f0 : fb0;
        float a1 = (kk + 1 < 64) ? f1 : fb1;
        float b0 = W[(n0 + ml) * K + kk];
        float b1 = W[(n0 + ml) * K + kk + 1];
        ua.p[j] = __builtin_amdgcn_cvt_pkrtz(a0, a1);
        ub.p[j] = __builtin_amdgcn_cvt_pkrtz(b0, b1);
      }
      acc = __builtin_amdgcn_wmma_f32_16x16x32_f16(false, ua.v, false, ub.v, (short)0, acc, false, false);
    }
#pragma unroll
    for (int r = 0; r < 8; ++r)
      tile[hf * 8 + r][n0 + ml] = acc[r] + bias[n0 + ml];
  }
  __syncthreads();
  if (tid < 16) {
    float s1 = 0.f, s2 = 0.f;
    for (int c = 0; c < 64; ++c) { float v = tile[tid][c]; s1 += v; s2 += v * v; }
    float m_ = s1 / 64.f;
    mu[tid] = m_;
    iv[tid] = rsqrtf(s2 / 64.f - m_ * m_ + 1e-5f);
  }
  __syncthreads();
  for (int idx = tid; idx < 1024; idx += 128) {
    int r = idx >> 6, c = idx & 63;
    float v = (tile[r][c] - mu[r]) * iv[r] * g[c] + be[c];
    int m = m0 + r;
    int b = m / F, t = m - b * F;
    xacc[(b * 64 + c) * F + t] += v;
  }
}

// ---------------- features ----------------
__global__ void k_features(const float* __restrict__ spec,
                           const float* __restrict__ erb_fb,
                           const float* __restrict__ mu_in, const float* __restrict__ s_in,
                           const float* __restrict__ erb_buf_in, const float* __restrict__ dfc0_buf_in,
                           float* __restrict__ mu_out, float* __restrict__ s_out,
                           float* __restrict__ e0in, float* __restrict__ c0in,
                           float* __restrict__ erb_buf_out, float* __restrict__ dfc0_buf_out)
{
  __shared__ float mag2[257];
  const int b = blockIdx.x, tid = threadIdx.x;
  for (int f = tid; f < 257; f += blockDim.x) {
    float re = spec[(b * 257 + f) * 2], im = spec[(b * 257 + f) * 2 + 1];
    mag2[f] = re * re + im * im + 1e-12f;
  }
  __syncthreads();
  if (tid < 32) {
    float acc = 0.f;
    for (int f = 0; f < 257; ++f) acc += mag2[f] * erb_fb[f * 32 + tid];
    float ferb = 10.f * log10f(acc + 1e-10f);
    float mu = 0.99f * mu_in[b * 32 + tid] + 0.01f * ferb;
    mu_out[b * 32 + tid] = mu;
    float fn = (ferb - mu) / (40.f + 1e-5f);
    e0in[b * 64 + tid]      = erb_buf_in[b * 32 + tid];
    e0in[b * 64 + 32 + tid] = fn;
    erb_buf_out[b * 32 + tid] = fn;
  }
  if (tid < 96) {
    float re = spec[(b * 257 + tid) * 2], im = spec[(b * 257 + tid) * 2 + 1];
    float xab = sqrtf(re * re + im * im + 1e-12f);
    float s = 0.99f * s_in[b * 96 + tid] + 0.01f * xab;
    s_out[b * 96 + tid] = s;
    float d = sqrtf(s + 1e-5f);
    float r = re / d, i = im / d;
    int cb = b * 384;
    c0in[cb + 0 * 96 + tid] = dfc0_buf_in[b * 192 + tid];
    c0in[cb + 1 * 96 + tid] = r;
    c0in[cb + 2 * 96 + tid] = dfc0_buf_in[b * 192 + 96 + tid];
    c0in[cb + 3 * 96 + tid] = i;
    dfc0_buf_out[b * 192 + tid]      = r;
    dfc0_buf_out[b * 192 + 96 + tid] = i;
  }
}

// ---------------- elementwise helpers ----------------
__global__ void k_copy(const float* __restrict__ a, float* __restrict__ o, int n) {
  int i = blockIdx.x * blockDim.x + threadIdx.x;
  if (i < n) o[i] = a[i];
}

// c1d (B,64,48) -> (B*48, 64) row-major
__global__ void k_tr48(const float* __restrict__ src, float* __restrict__ o) {
  int i = blockIdx.x * blockDim.x + threadIdx.x;
  if (i >= BATCH * 48 * 64) return;
  int m = i >> 6, c = i & 63;
  int b = m / 48, t = m - b * 48;
  o[i] = src[b * 3072 + c * 48 + t];
}

__global__ void k_emb0(const float* __restrict__ e3d, const float* __restrict__ cemb, float* __restrict__ o) {
  int i = blockIdx.x * blockDim.x + threadIdx.x;
  if (i >= BATCH * 512) return;
  int b = i >> 9, kk = i & 511;
  int f = kk >> 6, c = kk & 63;
  o[i] = e3d[b * 512 + c * 8 + f] + cemb[i];
}

__global__ void k_embdec_t(const float* __restrict__ e, float* __restrict__ o) {
  int i = blockIdx.x * blockDim.x + threadIdx.x;
  if (i >= BATCH * 512) return;
  int b = i >> 9, r = i & 511;
  int c = r >> 3, f = r & 7;
  o[i] = e[b * 512 + f * 64 + c];
}

__global__ void k_c0path(const float* __restrict__ dfpo, float* __restrict__ o) {
  int i = blockIdx.x * blockDim.x + threadIdx.x;
  if (i >= BATCH * 960) return;
  int b = i / 960, r = i - b * 960;
  int f = r / 10, oo = r - f * 10;
  o[i] = dfpo[(b * 10 + oo) * 96 + f];
}

__global__ void k_lsnr(const float* __restrict__ emb, const float* __restrict__ w,
                       const float* __restrict__ bb, float* __restrict__ o) {
  __shared__ float red[128];
  int b = blockIdx.x, tid = threadIdx.x;
  float s = 0.f;
  for (int k = tid; k < 512; k += 128) s += emb[b * 512 + k] * w[k];
  red[tid] = s; __syncthreads();
  for (int st = 64; st > 0; st >>= 1) { if (tid < st) red[tid] += red[tid + st]; __syncthreads(); }
  if (tid == 0) o[b] = sigf(red[0] + bb[0]) * 50.f - 15.f;
}

__global__ void k_window(const float* __restrict__ spec, const float* __restrict__ mask,
                         const float* __restrict__ dsb, float* __restrict__ win,
                         float* __restrict__ dsb_out) {
  int i = blockIdx.x * blockDim.x + threadIdx.x;
  if (i >= BATCH * 257) return;
  int b = i / 257, f = i - b * 257;
  float mf = mask[i];
  float sr = spec[i * 2] * mf, si = spec[i * 2 + 1] * mf;
  for (int t = 0; t < 4; ++t) {
    float vr = dsb[((b * 4 + t) * 257 + f) * 2];
    float vi = dsb[((b * 4 + t) * 257 + f) * 2 + 1];
    win[((b * 5 + t) * 257 + f) * 2]     = vr;
    win[((b * 5 + t) * 257 + f) * 2 + 1] = vi;
    if (t >= 1) {
      dsb_out[((b * 4 + t - 1) * 257 + f) * 2]     = vr;
      dsb_out[((b * 4 + t - 1) * 257 + f) * 2 + 1] = vi;
    }
  }
  win[((b * 5 + 4) * 257 + f) * 2]     = sr;
  win[((b * 5 + 4) * 257 + f) * 2 + 1] = si;
  dsb_out[((b * 4 + 3) * 257 + f) * 2]     = sr;
  dsb_out[((b * 4 + 3) * 257 + f) * 2 + 1] = si;
}

__global__ void k_final(const float* __restrict__ win, const float* __restrict__ coef,
                        float* __restrict__ se) {
  int i = blockIdx.x * blockDim.x + threadIdx.x;
  if (i >= BATCH * 257) return;
  int b = i / 257, f = i - b * 257;
  if (f < 96) {
    float fr = 0.f, fi = 0.f;
#pragma unroll
    for (int t = 0; t < 5; ++t) {
      float wr = win[((b * 5 + t) * 257 + f) * 2];
      float wi = win[((b * 5 + t) * 257 + f) * 2 + 1];
      float cr = coef[b * 960 + f * 10 + 2 * t];
      float ci = coef[b * 960 + f * 10 + 2 * t + 1];
      fr += wr * cr - wi * ci;
      fi += wr * ci + wi * cr;
    }
    se[i * 2] = fr; se[i * 2 + 1] = fi;
  } else {
    se[i * 2]     = win[((b * 5 + 4) * 257 + f) * 2];
    se[i * 2 + 1] = win[((b * 5 + 4) * 257 + f) * 2 + 1];
  }
}

// ================= host orchestration =================
struct GruW { const float *wih, *whh, *bih, *bhh; };
struct Blk  { GruW gf, gb; const float *fiw, *fib, *lnig, *lnib; GruW gt;
              const float *ftw, *ftb, *lntg, *lntb; };
struct Dp   { const float *inw, *inb, *outw, *outb; Blk blk[2]; };

extern "C" void kernel_launch(void* const* d_in, const int* in_sizes, int n_in,
                              void* d_out, int out_size, void* d_ws, size_t ws_size,
                              hipStream_t stream) {
  (void)in_sizes; (void)n_in; (void)out_size; (void)ws_size;
  // ---- inputs ----
  const float* spec        = (const float*)d_in[0];
  const float* erb_mu_in   = (const float*)d_in[1];
  const float* s_in        = (const float*)d_in[2];
  const float* enc_h_in    = (const float*)d_in[3];
  const float* erbdec_h_in = (const float*)d_in[4];
  const float* dfdec_h_in  = (const float*)d_in[5];
  const float* dst_e0      = (const float*)d_in[6];
  const float* dst_e1      = (const float*)d_in[7];
  const float* dst_d0      = (const float*)d_in[8];
  const float* dst_d1      = (const float*)d_in[9];
  const float* erb_c0_buf  = (const float*)d_in[10];
  const float* df_c0_buf   = (const float*)d_in[11];
  const float* df_cp_buf   = (const float*)d_in[12];
  const float* df_spec_buf = (const float*)d_in[13];
  int pi = 14;
  auto nx = [&]() { return (const float*)d_in[pi++]; };
  auto rdGru = [&]() { GruW g; g.wih = nx(); g.whh = nx(); g.bih = nx(); g.bhh = nx(); return g; };
  auto rdDp  = [&]() {
    Dp d; d.inw = nx(); d.inb = nx(); d.outw = nx(); d.outb = nx();
    for (int i = 0; i < 2; ++i) {
      Blk b; b.gf = rdGru(); b.gb = rdGru();
      b.fiw = nx(); b.fib = nx(); b.lnig = nx(); b.lnib = nx();
      b.gt = rdGru();
      b.ftw = nx(); b.ftb = nx(); b.lntg = nx(); b.lntb = nx();
      d.blk[i] = b;
    }
    return d;
  };
  const float* P_erb_fb  = nx();
  const float* P_erb_inv = nx();
  const float *ec0w = nx(), *ec0b = nx(), *ec1w = nx(), *ec1b = nx();
  const float *ec2w = nx(), *ec2b = nx(), *ec3w = nx(), *ec3b = nx();
  const float *dc0w = nx(), *dc0b = nx(), *dc1w = nx(), *dc1b = nx();
  Dp dpe = rdDp();
  Dp dpd = rdDp();
  const float *dffc_w = nx(), *dffc_b = nx();
  const float *eg_inw = nx(), *eg_inb = nx(); GruW eg_g = rdGru();
  const float *eg_outw = nx(), *eg_outb = nx(), *eg_skw = nx(), *eg_skb = nx();
  const float *lsnr_w = nx(), *lsnr_b = nx();
  const float *rg_inw = nx(), *rg_inb = nx(); GruW rg_g = rdGru();
  const float *rg_outw = nx(), *rg_outb = nx(), *rg_skw = nx(), *rg_skb = nx();
  const float *c3pw = nx(), *c3pb = nx(), *ct3w = nx(), *ct3b = nx();
  const float *c2pw = nx(), *c2pb = nx(), *ct2w = nx(), *ct2b = nx();
  const float *c1pw = nx(), *c1pb = nx(), *ct1w = nx(), *ct1b = nx();
  const float *c0pw = nx(), *c0pb = nx(), *coutw = nx(), *coutb = nx();
  const float *fg_inw = nx(), *fg_inb = nx(); GruW fg_g = rdGru();
  const float *fg_outw = nx(), *fg_outb = nx();
  const float *dfskw = nx(), *dfskb = nx();
  const float *dfpw = nx(), *dfpb = nx();
  const float *dfow = nx(), *dfob = nx();

  // ---- outputs (concatenated in return order) ----
  float* ob = (float*)d_out; size_t oo = 0;
  auto OA = [&](size_t n) { float* p = ob + oo; oo += n; return p; };
  float* o_spec_e    = OA((size_t)256 * 257 * 2);
  float* o_lsnr      = OA(256);
  float* o_mu        = OA(256 * 32);
  float* o_s         = OA(256 * 96);
  float* o_ench      = OA(256 * 256);
  float* o_erbh      = OA(256 * 256);
  float* o_dfh       = OA(256 * 256);
  float* o_ste0      = OA((size_t)256 * 8 * 64);
  float* o_ste1      = OA((size_t)256 * 8 * 64);
  float* o_std0      = OA((size_t)256 * 48 * 64);
  float* o_std1      = OA((size_t)256 * 48 * 64);
  float* o_erbbuf    = OA(256 * 32);
  float* o_dfc0buf   = OA((size_t)256 * 2 * 96);
  float* o_dfcpbuf   = OA((size_t)256 * 64 * 96);
  float* o_dfspecbuf = OA((size_t)256 * 4 * 257 * 2);

  // ---- workspace bump allocator ----
  float* wb = (float*)d_ws; size_t wo = 0;
  auto WA = [&](size_t n) { float* p = wb + wo; wo += n; return p; };
  float* ws_e0in  = WA((size_t)256 * 64);
  float* ws_c0in  = WA((size_t)256 * 384);
  float* ws_e0    = WA((size_t)256 * 64 * 32);
  float* ws_e1    = WA((size_t)256 * 64 * 16);
  float* ws_e2    = WA((size_t)256 * 64 * 8);
  float* ws_e3    = WA((size_t)256 * 64 * 8);
  float* ws_e3d   = WA((size_t)256 * 64 * 8);
  float* ws_c0    = WA((size_t)256 * 64 * 96);
  float* ws_c1    = WA((size_t)256 * 64 * 48);
  float* ws_c1d   = WA((size_t)256 * 64 * 48);
  float* ws_c1dt  = WA((size_t)256 * 48 * 64);
  float* ws_dpx8  = WA((size_t)256 * 64 * 8);
  float* ws_dpx48 = WA((size_t)256 * 64 * 48);
  float* ws_fwd   = WA((size_t)256 * 48 * 64);
  float* ws_bwd   = WA((size_t)256 * 48 * 64);
  float* ws_hseq  = WA((size_t)256 * 48 * 64);
  float* ws_cemb  = WA((size_t)256 * 512);
  float* ws_emb0  = WA((size_t)256 * 512);
  float* ws_xi    = WA((size_t)256 * 256);
  float* ws_h1    = WA((size_t)256 * 256);
  float* ws_skipb = WA((size_t)256 * 512);
  float* ws_emb   = WA((size_t)256 * 512);
  float* ws_emdec = WA((size_t)256 * 512);
  float* ws_emdt  = WA((size_t)256 * 512);
  float* ws_t3    = WA((size_t)256 * 64 * 8);
  float* ws_d3    = WA((size_t)256 * 64 * 8);
  float* ws_t2    = WA((size_t)256 * 64 * 8);
  float* ws_d2    = WA((size_t)256 * 64 * 16);
  float* ws_t1    = WA((size_t)256 * 64 * 16);
  float* ws_d1    = WA((size_t)256 * 64 * 32);
  float* ws_t0    = WA((size_t)256 * 64 * 32);
  float* ws_m     = WA((size_t)256 * 32);
  float* ws_mask  = WA((size_t)256 * 257);
  float* ws_win   = WA((size_t)256 * 5 * 257 * 2);
  float* ws_cdf   = WA((size_t)256 * 256);
  float* ws_dfpo  = WA((size_t)256 * 10 * 96);
  float* ws_c0p   = WA((size_t)256 * 960);
  float* ws_coef  = WA((size_t)256 * 960);

  auto gemm = [&](const float* A, int lda, const float* W, int wsn, int wsk,
                  const float* bias, const float* res, int res_sm,
                  float* out, int out_sm, int M, int N, int K, int act) {
    int tiles = (N + 15) / 16;
    dim3 grid(M / 16, (tiles + 7) / 8);
    k_gemm<<<grid, 256, 0, stream>>>(A, lda, W, wsn, wsk, bias, res, res_sm, out, out_sm, N, K, act);
  };
  auto conv = [&](const float* in, const float* in2, int Cin, int Fin, const float* W,
                  const float* bias, const float* res, float* out, int Cout, int Fout,
                  int KW, int strd, int pad, int mode, int act) {
    dim3 grid(16, Fout);
    if (KW == 1)
      k_conv<1, 0, 0><<<grid, 128, 0, stream>>>(in, in2, Cin, Fin, W, bias, res, out, Cout, Fout, strd, pad, act);
    else if (mode == 1)
      k_conv<3, 1, 0><<<grid, 128, 0, stream>>>(in, in2, Cin, Fin, W, bias, res, out, Cout, Fout, strd, pad, act);
    else if (in2)
      k_conv<3, 0, 1><<<grid, 128, 0, stream>>>(in, in2, Cin, Fin, W, bias, res, out, Cout, Fout, strd, pad, act);
    else
      k_conv<3, 0, 0><<<grid, 128, 0, stream>>>(in, in2, Cin, Fin, W, bias, res, out, Cout, Fout, strd, pad, act);
  };
  auto dprnn = [&](const float* xin, float* xout, int F, Dp& p,
                   const float* h0in, float* h0out, const float* h1in, float* h1out,
                   float* dpx) {
    conv(xin, nullptr, 64, F, p.inw, p.inb, nullptr, dpx, 64, F, 1, 1, 0, 0, 0);
    const float* hins[2]  = { h0in, h1in };
    float*       houts[2] = { h0out, h1out };
    for (int i = 0; i < 2; ++i) {
      Blk& bk = p.blk[i];
      k_gru_seq<<<16, 128, 0, stream>>>(dpx, F, bk.gf.wih, bk.gf.whh, bk.gf.bih, bk.gf.bhh, ws_fwd, 0, 1);
      k_gru_seq<<<16, 128, 0, stream>>>(dpx, F, bk.gb.wih, bk.gb.whh, bk.gb.bih, bk.gb.bhh, ws_bwd, F - 1, -1);
      int M = 256 * F;
      k_lin_ln_add<<<M / 16, 128, 0, stream>>>(ws_fwd, ws_bwd, 128, bk.fiw, bk.fib,
                                               bk.lnig, bk.lnib, dpx, F);
      // x-gather: idx = (m/F)*64F + (m%F) + kk*F
      k_gru_cell<<<M / 16, 256, 2 * 16 * 192 * sizeof(float), stream>>>(
          dpx, F, 64 * F, 1, F, hins[i], bk.gt.wih, bk.gt.whh, bk.gt.bih, bk.gt.bhh,
          houts[i], ws_hseq, 64, 6);
      k_lin_ln_add<<<M / 16, 128, 0, stream>>>(ws_hseq, ws_hseq, 64, bk.ftw, bk.ftb,
                                               bk.lntg, bk.lntb, dpx, F);
    }
    conv(dpx, nullptr, 64, F, p.outw, p.outb, nullptr, xout, 64, F, 1, 1, 0, 0, 0);
  };
  auto sgru = [&](const float* inp, int Kin, const float* in_w, const float* in_b,
                  GruW& g, const float* out_w, const float* out_b,
                  const float* skip_w, const float* skip_b, int Nout,
                  const float* h_in, float* h_out, float* outbuf) {
    gemm(inp, Kin, in_w, Kin, 1, in_b, nullptr, 0, ws_xi, 256, 256, 256, Kin, 1);
    k_gru_cell<<<16, 256, 2 * 16 * 768 * sizeof(float), stream>>>(
        ws_xi, 1, 256, 0, 1, h_in, g.wih, g.whh, g.bih, g.bhh, h_out, ws_h1, 256, 8);
    const float* res = nullptr;
    if (skip_w) {
      gemm(inp, Kin, skip_w, Kin, 1, skip_b, nullptr, 0, ws_skipb, Nout, 256, Nout, Kin, 0);
      res = ws_skipb;
    }
    gemm(ws_h1, 256, out_w, 256, 1, out_b, res, Nout, outbuf, Nout, 256, Nout, 256, 0);
  };

  // ---- pipeline ----
  k_features<<<256, 256, 0, stream>>>(spec, P_erb_fb, erb_mu_in, s_in, erb_c0_buf, df_c0_buf,
                                      o_mu, o_s, ws_e0in, ws_c0in, o_erbbuf, o_dfc0buf);
  // encoder (ERB branch)
  conv(ws_e0in, nullptr, 2, 32, ec0w, ec0b, nullptr, ws_e0, 64, 32, 3, 1, 1, 0, 1);
  conv(ws_e0, nullptr, 64, 32, ec1w, ec1b, nullptr, ws_e1, 64, 16, 3, 2, 1, 0, 1);
  conv(ws_e1, nullptr, 64, 16, ec2w, ec2b, nullptr, ws_e2, 64, 8, 3, 2, 1, 0, 1);
  conv(ws_e2, nullptr, 64, 8, ec3w, ec3b, nullptr, ws_e3, 64, 8, 3, 1, 1, 0, 1);
  dprnn(ws_e3, ws_e3d, 8, dpe, dst_e0, o_ste0, dst_e1, o_ste1, ws_dpx8);
  // encoder (DF branch)
  conv(ws_c0in, nullptr, 4, 96, dc0w, dc0b, nullptr, ws_c0, 64, 96, 3, 1, 1, 0, 1);
  k_copy<<<(256 * 64 * 96 + 255) / 256, 256, 0, stream>>>(ws_c0, o_dfcpbuf, 256 * 64 * 96);
  conv(ws_c0, nullptr, 64, 96, dc1w, dc1b, nullptr, ws_c1, 64, 48, 3, 2, 1, 0, 1);
  dprnn(ws_c1, ws_c1d, 48, dpd, dst_d0, o_std0, dst_d1, o_std1, ws_dpx48);
  // embedding
  k_tr48<<<(256 * 48 * 64 + 255) / 256, 256, 0, stream>>>(ws_c1d, ws_c1dt);
  gemm(ws_c1dt, 3072, dffc_w, 3072, 1, dffc_b, nullptr, 0, ws_cemb, 512, 256, 512, 3072, 1);
  k_emb0<<<512, 256, 0, stream>>>(ws_e3d, ws_cemb, ws_emb0);
  sgru(ws_emb0, 512, eg_inw, eg_inb, eg_g, eg_outw, eg_outb, eg_skw, eg_skb, 512,
       enc_h_in, o_ench, ws_emb);
  k_lsnr<<<256, 128, 0, stream>>>(ws_emb, lsnr_w, lsnr_b, o_lsnr);
  // ERB decoder
  sgru(ws_emb, 512, rg_inw, rg_inb, rg_g, rg_outw, rg_outb, rg_skw, rg_skb, 512,
       erbdec_h_in, o_erbh, ws_emdec);
  k_embdec_t<<<512, 256, 0, stream>>>(ws_emdec, ws_emdt);
  conv(ws_e3, nullptr, 64, 8, c3pw, c3pb, ws_emdt, ws_t3, 64, 8, 1, 1, 0, 0, 0);
  conv(ws_t3, nullptr, 64, 8, ct3w, ct3b, nullptr, ws_d3, 64, 8, 3, 1, 1, 0, 1);
  conv(ws_e2, nullptr, 64, 8, c2pw, c2pb, ws_d3, ws_t2, 64, 8, 1, 1, 0, 0, 0);
  conv(ws_t2, nullptr, 64, 8, ct2w, ct2b, nullptr, ws_d2, 64, 16, 3, 1, 1, 1, 1);
  conv(ws_e1, nullptr, 64, 16, c1pw, c1pb, ws_d2, ws_t1, 64, 16, 1, 1, 0, 0, 0);
  conv(ws_t1, nullptr, 64, 16, ct1w, ct1b, nullptr, ws_d1, 64, 32, 3, 1, 1, 1, 1);
  conv(ws_e0, nullptr, 64, 32, c0pw, c0pb, ws_d1, ws_t0, 64, 32, 1, 1, 0, 0, 0);
  conv(ws_t0, nullptr, 64, 32, coutw, coutb, nullptr, ws_m, 1, 32, 3, 1, 1, 0, 2);
  gemm(ws_m, 32, P_erb_inv, 1, 257, nullptr, nullptr, 0, ws_mask, 257, 256, 257, 32, 0);
  k_window<<<(256 * 257 + 255) / 256, 256, 0, stream>>>(spec, ws_mask, df_spec_buf, ws_win, o_dfspecbuf);
  // DF decoder
  sgru(ws_emb, 512, fg_inw, fg_inb, fg_g, fg_outw, fg_outb, nullptr, nullptr, 256,
       dfdec_h_in, o_dfh, ws_cdf);
  gemm(ws_emb, 512, dfskw, 512, 1, dfskb, ws_cdf, 256, ws_cdf, 256, 256, 256, 512, 0);
  conv(df_cp_buf, ws_c0, 128, 96, dfpw, dfpb, nullptr, ws_dfpo, 10, 96, 3, 1, 1, 0, 0);
  k_c0path<<<(256 * 960 + 255) / 256, 256, 0, stream>>>(ws_dfpo, ws_c0p);
  gemm(ws_cdf, 256, dfow, 256, 1, dfob, ws_c0p, 960, ws_coef, 960, 256, 960, 256, 3);
  k_final<<<(256 * 257 + 255) / 256, 256, 0, stream>>>(ws_win, ws_coef, o_spec_e);
}
